// causality_enhanced_hetero_graph_model_24489903522213
// MI455X (gfx1250) — compile-verified, matrix operand order checked by reading, not run on hardware
//
#include <hip/hip_runtime.h>
#include <hip/hip_bf16.h>
#include <math.h>

// ---------------------------------------------------------------------------
// CDNA5 (gfx1250) HGT-style hetero-graph attention forward.
//   Stage 1: bf16 WMMA GEMMs for the 18 linear projections (wave32, 16x16 tiles,
//            v_wmma_f32_16x16x32_bf16, K=256 -> 8 WMMAs/tile, f32 accumulate)
//   Stage 2: per-edge attention (VALU) + WMMA-batched message einsum
//            (16 edges x 32k x 32d per wave, K=32 -> 2 WMMAs) + segment softmax
//            via f32 global atomics
//   Stage 3: mean over etypes -> ReLU -> LayerNorm (LDS block reduction)
// ---------------------------------------------------------------------------

typedef __bf16 bf16_t;
typedef __attribute__((ext_vector_type(16))) __bf16 bf16x16;
typedef __attribute__((ext_vector_type(8)))  __bf16 bf16x8;
typedef __attribute__((ext_vector_type(8)))  float  f32x8;

__device__ __forceinline__ float wave_sum32(float x) {
#pragma unroll
  for (int off = 16; off > 0; off >>= 1) x += __shfl_xor(x, off, 32);
  return x;
}

__device__ __forceinline__ void atomic_max_f32(float* addr, float v) {
  unsigned int* u = (unsigned int*)addr;
  unsigned int cur = __float_as_uint(*(volatile float*)addr);
  while (__uint_as_float(cur) < v) {
    unsigned int prev = atomicCAS(u, cur, __float_as_uint(v));
    if (prev == cur) break;
    cur = prev;
  }
}

// ------------------------------ utility kernels ----------------------------

__global__ __launch_bounds__(256) void hgt_fill_f32(float* p, float v, int n) {
  int i = blockIdx.x * blockDim.x + threadIdx.x;
  if (i < n) p[i] = v;
}

__global__ __launch_bounds__(256) void hgt_cvt_bf16(const float* __restrict__ s,
                                                    bf16_t* __restrict__ d, int n) {
  int i = blockIdx.x * blockDim.x + threadIdx.x;
  if (i < n) d[i] = (bf16_t)s[i];
}

// d[i] = bf16(s[i] + te[i&31])   (CV + time_emb, broadcast over heads)
__global__ __launch_bounds__(256) void hgt_cvt_te_bf16(const float* __restrict__ s,
                                                       const float* __restrict__ te,
                                                       bf16_t* __restrict__ d, int n) {
  int i = blockIdx.x * blockDim.x + threadIdx.x;
  if (i < n) d[i] = (bf16_t)(s[i] + te[i & 31]);
}

// W [256,256] row-major (k,n) -> Wt[n][k] bf16 so WMMA B loads are contiguous.
__global__ __launch_bounds__(256) void hgt_transpose_w(const float* __restrict__ W,
                                                       bf16_t* __restrict__ Wt) {
  int i = blockIdx.x * blockDim.x + threadIdx.x;  // 65536 elements
  int n = i >> 8, k = i & 255;
  Wt[n * 256 + k] = (bf16_t)W[k * 256 + n];
}

// rel_msg f32 [8,32(k),32(d)] -> bf16 [8,32(d),32(k)] (B-operand friendly)
__global__ __launch_bounds__(256) void hgt_transpose_msg(const float* __restrict__ M,
                                                         bf16_t* __restrict__ Mt) {
  int i = blockIdx.x * blockDim.x + threadIdx.x;  // 8192 elements
  if (i >= 8192) return;
  int h = i >> 10, rem = i & 1023;
  int d = rem >> 5, k = rem & 31;
  Mt[i] = (bf16_t)M[h * 1024 + k * 32 + d];
}

// ------------------------------ WMMA GEMM ----------------------------------
// out[M,256] = A[M,256](bf16) @ W[256,256] + bias ; Wt is W transposed [N,K].
// One wave per 16x16 tile. Per the 16-bit A/B layout: lane (m|n = lane&15,
// kh = lane>>4) holds K = kh*8..kh*8+7 (vec elems 0..7) and +16 (elems 8..15).
__global__ __launch_bounds__(256) void hgt_gemm_bf16_wmma(
    const bf16_t* __restrict__ A, const bf16_t* __restrict__ Wt,
    const float* __restrict__ bias, float* __restrict__ out, int M) {
  const int wave = (int)((blockIdx.x * blockDim.x + threadIdx.x) >> 5);
  const int lane = (int)(threadIdx.x & 31);
  const int mtiles = (M + 15) >> 4;
  if (wave >= mtiles * 16) return;
  const int mt = wave >> 4;
  const int nt = wave & 15;
  const int m0 = mt << 4;
  const int kh = lane >> 4;
  const int l16 = lane & 15;
  int arow_i = m0 + l16;
  if (arow_i >= M) arow_i = 0;  // A row m only affects C row m; guarded on store
  const bf16_t* __restrict__ arow = A + (size_t)arow_i * 256;
  const bf16_t* __restrict__ brow = Wt + (size_t)((nt << 4) + l16) * 256;
  f32x8 c = {};
#pragma unroll
  for (int ks = 0; ks < 8; ++ks) {
    const int kb = ks * 32 + kh * 8;
    union { bf16x16 v; bf16x8 h[2]; } a, b;
    a.h[0] = *(const bf16x8*)(arow + kb);
    a.h[1] = *(const bf16x8*)(arow + kb + 16);
    b.h[0] = *(const bf16x8*)(brow + kb);
    b.h[1] = *(const bf16x8*)(brow + kb + 16);
    c = __builtin_amdgcn_wmma_f32_16x16x32_bf16(false, a.v, false, b.v,
                                                (short)0, c, false, false);
  }
  const int ni = (nt << 4) + l16;
  const float bv = bias[ni];
#pragma unroll
  for (int r = 0; r < 8; ++r) {
    const int m = m0 + kh * 8 + r;  // C layout: lanes<16 -> M=r, lanes>=16 -> M=8+r
    if (m < M) out[(size_t)m * 256 + ni] = c[r] + bv;
  }
}

// ------------------------------ edge kernels -------------------------------
// att[e,h] = sum_k (Q[dst]+w*rel_att)[h,k] * K[src][h,k] / sqrt(32); track seg max.
__global__ __launch_bounds__(256) void hgt_edge_att(
    const float* __restrict__ Qd, const float* __restrict__ Ksrc,
    const int* __restrict__ src, const int* __restrict__ dst,
    const float* __restrict__ ew, const float* __restrict__ rel_att,
    float* __restrict__ att, float* __restrict__ dmax, int E) {
  const int e = (int)((blockIdx.x * blockDim.x + threadIdx.x) >> 5);
  const int lane = (int)(threadIdx.x & 31);
  if (e >= E) return;
  const int s = src[e], d = dst[e];
  const float w = ew[e];
  const float* __restrict__ q = Qd + (size_t)d * 256;
  const float* __restrict__ k = Ksrc + (size_t)s * 256;
  const float rinv = 0.17677669529663687f;  // 1/sqrt(32)
#pragma unroll
  for (int h = 0; h < 8; ++h) {
    float t = (q[h * 32 + lane] + w * rel_att[h * 32 + lane]) * k[h * 32 + lane];
    float a = wave_sum32(t) * rinv;
    if (lane == 0) {
      att[(size_t)e * 8 + h] = a;
      atomic_max_f32(&dmax[(size_t)d * 8 + h], a);
    }
  }
}

// catt[e,h] = sum_d CQ[dst][h,d] * (sum_k CK[src][h,k]*F[c,h,k,d]) / sqrt(32)
// (cau_filter varies per edge via cau_type[src] -> no shared B tile, keep VALU)
__global__ __launch_bounds__(256) void hgt_edge_catt(
    const float* __restrict__ CQd, const float* __restrict__ CKsrc,
    const int* __restrict__ src, const int* __restrict__ dst,
    const int* __restrict__ cau_type, const float* __restrict__ cau_filter,
    float* __restrict__ att, float* __restrict__ dmax, int E) {
  const int e = (int)((blockIdx.x * blockDim.x + threadIdx.x) >> 5);
  const int lane = (int)(threadIdx.x & 31);
  if (e >= E) return;
  const int s = src[e], d = dst[e];
  const int c = cau_type[s];
  const float* __restrict__ ck = CKsrc + (size_t)s * 256;
  const float* __restrict__ cq = CQd + (size_t)d * 256;
  const float rinv = 0.17677669529663687f;
  for (int h = 0; h < 8; ++h) {
    const float ckl = ck[h * 32 + lane];
    const float* __restrict__ F = cau_filter + (size_t)((c * 8 + h) * 32) * 32;
    float tmp = 0.f;
#pragma unroll
    for (int kk = 0; kk < 32; ++kk)
      tmp += __shfl(ckl, kk, 32) * F[kk * 32 + lane];  // masked[d=lane]
    float a = wave_sum32(tmp * cq[h * 32 + lane]) * rinv;
    if (lane == 0) {
      att[(size_t)e * 8 + h] = a;
      atomic_max_f32(&dmax[(size_t)d * 8 + h], a);
    }
  }
}

// p = exp(att - segmax) in place; psum[dst,h] += p. One thread per (e,h).
__global__ __launch_bounds__(256) void hgt_edge_p(
    const int* __restrict__ dst, const float* __restrict__ dmax,
    float* __restrict__ att_p, float* __restrict__ psum, int E) {
  int i = blockIdx.x * blockDim.x + threadIdx.x;
  if (i >= E * 8) return;
  int e = i >> 3, h = i & 7;
  int d = dst[e];
  float p = __expf(att_p[i] - dmax[(size_t)d * 8 + h]);
  att_p[i] = p;
  atomicAdd(&psum[(size_t)d * 8 + h], p);
}

// WMMA-batched message einsum: one wave = 16 edges x 1 head.
// A tile: 16 gathered V rows (bf16) x K=32 ; B: msgT[h] (d-major) -> 2 WMMAs
// covering d=0..15 and d=16..31; then hacc[dst,h,d] += p[e,h] * val.
__global__ __launch_bounds__(256) void hgt_edge_msg_wmma(
    const bf16_t* __restrict__ Vbf, const int* __restrict__ src,
    const int* __restrict__ dst, const bf16_t* __restrict__ msgT,
    const float* __restrict__ p, float* __restrict__ hacc, int E) {
  const int wave = (int)((blockIdx.x * blockDim.x + threadIdx.x) >> 5);
  const int lane = (int)(threadIdx.x & 31);
  const int groups = (E + 15) >> 4;
  if (wave >= groups * 8) return;
  const int g = wave >> 3;
  const int h = wave & 7;
  const int l16 = lane & 15;
  const int kh = lane >> 4;
  int ea = g * 16 + l16;
  if (ea >= E) ea = E - 1;  // duplicate row; per-row stores guarded below
  const bf16_t* __restrict__ arow = Vbf + (size_t)src[ea] * 256 + h * 32;
  union { bf16x16 v; bf16x8 x[2]; } a, b0, b1;
  a.x[0] = *(const bf16x8*)(arow + kh * 8);
  a.x[1] = *(const bf16x8*)(arow + kh * 8 + 16);
  const bf16_t* __restrict__ m0 = msgT + ((size_t)h * 32 + l16) * 32;       // d=l16
  const bf16_t* __restrict__ m1 = msgT + ((size_t)h * 32 + 16 + l16) * 32;  // d=16+l16
  b0.x[0] = *(const bf16x8*)(m0 + kh * 8);
  b0.x[1] = *(const bf16x8*)(m0 + kh * 8 + 16);
  b1.x[0] = *(const bf16x8*)(m1 + kh * 8);
  b1.x[1] = *(const bf16x8*)(m1 + kh * 8 + 16);
  f32x8 c0 = {}, c1 = {};
  c0 = __builtin_amdgcn_wmma_f32_16x16x32_bf16(false, a.v, false, b0.v,
                                               (short)0, c0, false, false);
  c1 = __builtin_amdgcn_wmma_f32_16x16x32_bf16(false, a.v, false, b1.v,
                                               (short)0, c1, false, false);
#pragma unroll
  for (int r = 0; r < 8; ++r) {
    const int e = g * 16 + kh * 8 + r;  // C row r (+8 for upper half) = edge
    if (e < E) {
      const float pe = p[(size_t)e * 8 + h];
      float* __restrict__ hrow = hacc + (size_t)dst[e] * 256 + h * 32;
      atomicAdd(hrow + l16, pe * c0[r]);
      atomicAdd(hrow + 16 + l16, pe * c1[r]);
    }
  }
}

// out_acc += where(psum>0, hacc/max(psum,1e-9), 0)
__global__ __launch_bounds__(256) void hgt_norm_acc(
    const float* __restrict__ hacc, const float* __restrict__ psum,
    float* __restrict__ out_acc, int N) {
  int i = blockIdx.x * blockDim.x + threadIdx.x;
  if (i >= N * 256) return;
  int n = i >> 8;
  int h = (i >> 5) & 7;
  float s = psum[(size_t)n * 8 + h];
  out_acc[i] += (s > 0.f) ? hacc[i] / fmaxf(s, 1e-9f) : 0.f;
}

// x = relu(acc/cnt); LayerNorm over 256 with LDS tree reduction.
__global__ __launch_bounds__(256) void hgt_finalize(
    const float* __restrict__ out_acc, const float* __restrict__ g,
    const float* __restrict__ b, float* __restrict__ out, float inv_cnt, int N) {
  __shared__ float red[256];
  const int n = blockIdx.x;
  const int t = threadIdx.x;
  if (n >= N) return;
  float x = fmaxf(out_acc[(size_t)n * 256 + t] * inv_cnt, 0.f);
  red[t] = x;
  __syncthreads();
  for (int s = 128; s > 0; s >>= 1) {
    if (t < s) red[t] += red[t + s];
    __syncthreads();
  }
  float mu = red[0] * (1.f / 256.f);
  __syncthreads();
  float dx = x - mu;
  red[t] = dx * dx;
  __syncthreads();
  for (int s = 128; s > 0; s >>= 1) {
    if (t < s) red[t] += red[t + s];
    __syncthreads();
  }
  float var = red[0] * (1.f / 256.f);
  out[(size_t)n * 256 + t] = dx * rsqrtf(var + 1e-5f) * g[t] + b[t];
}

// ------------------------------ host launch --------------------------------
// Flattened input order (JAX sorted-dict tree flatten):
//  0 cau_type | 1..6 edge_weight(dw,td,tt,tw,wd,ww) | 7..18 edges src/dst pairs
//  19 h_doc 20 h_topic 21 h_word | 22 cau_filter
//  23..34 lin_doc (ck_b,ck_w,cq_b,cq_w,cv_b,cv_w,k_b,k_w,q_b,q_w,v_b,v_w)
//  35..46 lin_topic | 47..58 lin_word
//  59..64 ln (doc b,g / topic b,g / word b,g)
//  65..70 rel_att(dw..ww) | 71..76 rel_msg(dw..ww) | 77..79 rel_msg_cau(td,tt,tw)
//  80 time_emb
extern "C" void kernel_launch(void* const* d_in, const int* in_sizes, int n_in,
                              void* d_out, int out_size, void* d_ws, size_t ws_size,
                              hipStream_t stream) {
  (void)in_sizes; (void)n_in; (void)out_size; (void)ws_size;
  const int NW = 30000, NT = 1000, ND = 5000;
  const int NN_[3] = {NW, NT, ND};              // WORD=0, TOPIC=1, DOC=2
  const int feat_idx[3] = {21, 20, 19};
  const int lin_base[3] = {47, 35, 23};
  const int ln_b_idx[3] = {63, 61, 59};
  const int ln_g_idx[3] = {64, 62, 60};
  const size_t out_off[3] = {0, (size_t)NW * 256, (size_t)(NW + NT) * 256};
  enum { PCK = 0, PCQ = 1, PCV = 2, PK = 3, PQ = 4, PV = 5 };

  char* ws = (char*)d_ws;
  size_t off = 0;
  auto alloc = [&](size_t bytes) -> char* {
    char* p = ws + off;
    off += (bytes + 255) & ~(size_t)255;
    return p;
  };
  bf16_t* bfeat[3];
  for (int t = 0; t < 3; ++t) bfeat[t] = (bf16_t*)alloc((size_t)NN_[t] * 256 * 2);
  bf16_t* bwt[3][6];
  for (int t = 0; t < 3; ++t)
    for (int p = 0; p < 6; ++p) bwt[t][p] = (bf16_t*)alloc((size_t)65536 * 2);
  float* proj[3][6];
  for (int t = 0; t < 3; ++t)
    for (int p = 0; p < 6; ++p) proj[t][p] = (float*)alloc((size_t)NN_[t] * 256 * 4);
  bf16_t* vbf[3];
  for (int t = 0; t < 3; ++t) vbf[t] = (bf16_t*)alloc((size_t)NN_[t] * 256 * 2);
  bf16_t* cvbf = (bf16_t*)alloc((size_t)NT * 256 * 2);  // CV+time, topic only
  bf16_t* msgT = (bf16_t*)alloc((size_t)9 * 8192 * 2);  // 6 rel_msg + 3 causal
  float* att  = (float*)alloc((size_t)400000 * 8 * 4);
  float* dmax = (float*)alloc((size_t)NW * 8 * 4);
  float* psum = (float*)alloc((size_t)NW * 8 * 4);
  float* hacc = (float*)alloc((size_t)NW * 256 * 4);
  float* oacc = (float*)alloc((size_t)(NW + NT + ND) * 256 * 4);

  // Stage 0: bf16 conversion of features, transposed weights, message matrices
  for (int t = 0; t < 3; ++t) {
    int n = NN_[t] * 256;
    hgt_cvt_bf16<<<(n + 255) / 256, 256, 0, stream>>>((const float*)d_in[feat_idx[t]],
                                                      bfeat[t], n);
    for (int p = 0; p < 6; ++p)
      hgt_transpose_w<<<65536 / 256, 256, 0, stream>>>(
          (const float*)d_in[lin_base[t] + 2 * p + 1], bwt[t][p]);
  }
  for (int i = 0; i < 6; ++i)
    hgt_transpose_msg<<<8192 / 256, 256, 0, stream>>>((const float*)d_in[71 + i],
                                                      msgT + (size_t)i * 8192);
  for (int j = 0; j < 3; ++j)
    hgt_transpose_msg<<<8192 / 256, 256, 0, stream>>>((const float*)d_in[77 + j],
                                                      msgT + (size_t)(6 + j) * 8192);

  // Stage 1: 18 WMMA GEMM projections
  for (int t = 0; t < 3; ++t) {
    int M = NN_[t];
    int waves = ((M + 15) / 16) * 16;
    int blocks = (waves * 32 + 255) / 256;
    for (int p = 0; p < 6; ++p)
      hgt_gemm_bf16_wmma<<<blocks, 256, 0, stream>>>(
          bfeat[t], bwt[t][p], (const float*)d_in[lin_base[t] + 2 * p],
          proj[t][p], M);
  }

  // V projections -> bf16 (A operands of the message WMMA); CV+time for topic
  for (int t = 0; t < 3; ++t) {
    int n = NN_[t] * 256;
    hgt_cvt_bf16<<<(n + 255) / 256, 256, 0, stream>>>(proj[t][PV], vbf[t], n);
  }
  hgt_cvt_te_bf16<<<(NT * 256 + 255) / 256, 256, 0, stream>>>(
      proj[1][PCV], (const float*)d_in[80], cvbf, NT * 256);

  // zero global accumulator
  {
    int n = (NW + NT + ND) * 256;
    hgt_fill_f32<<<(n + 255) / 256, 256, 0, stream>>>(oacc, 0.f, n);
  }

  // Stage 2: per-etype attention rounds (sorted etype order: dw,td,tt,tw,wd,ww)
  struct ED { int ew, src, dst, E, st, dt, ra, rm, cm; };
  const ED ed[6] = {
      {1, 7, 8, 200000, 2, 0, 65, 0, -1},   // dw: doc -> word
      {2, 9, 10, 50000, 1, 2, 66, 1, 6},    // td: topic -> doc   (causal)
      {3, 11, 12, 30000, 1, 1, 67, 2, 7},   // tt: topic -> topic (causal)
      {4, 13, 14, 100000, 1, 0, 68, 3, 8},  // tw: topic -> word  (causal)
      {5, 15, 16, 200000, 0, 2, 69, 4, -1}, // wd: word -> doc
      {6, 17, 18, 400000, 0, 0, 70, 5, -1}, // ww: word -> word
  };
  const float NEGINF = -INFINITY;
  for (int i = 0; i < 6; ++i) {
    const ED& e = ed[i];
    const int Nd = NN_[e.dt];
    const int* srcp = (const int*)d_in[e.src];
    const int* dstp = (const int*)d_in[e.dst];
    const int eblocks = (e.E + 7) / 8;       // 8 waves (edges) per 256-thread block
    const int groups = (e.E + 15) / 16;      // msg WMMA: groups blocks of 8 waves
    for (int pass = 0; pass < (e.cm >= 0 ? 2 : 1); ++pass) {
      hgt_fill_f32<<<(Nd * 8 + 255) / 256, 256, 0, stream>>>(dmax, NEGINF, Nd * 8);
      hgt_fill_f32<<<(Nd * 8 + 255) / 256, 256, 0, stream>>>(psum, 0.f, Nd * 8);
      hgt_fill_f32<<<(Nd * 256 + 255) / 256, 256, 0, stream>>>(hacc, 0.f, Nd * 256);
      if (pass == 0) {
        hgt_edge_att<<<eblocks, 256, 0, stream>>>(
            proj[e.dt][PQ], proj[e.st][PK], srcp, dstp,
            (const float*)d_in[e.ew], (const float*)d_in[e.ra], att, dmax, e.E);
      } else {
        hgt_edge_catt<<<eblocks, 256, 0, stream>>>(
            proj[e.dt][PCQ], proj[e.st][PCK], srcp, dstp,
            (const int*)d_in[0], (const float*)d_in[22], att, dmax, e.E);
      }
      hgt_edge_p<<<(e.E * 8 + 255) / 256, 256, 0, stream>>>(dstp, dmax, att,
                                                            psum, e.E);
      const bf16_t* vsrc = (pass == 0) ? vbf[e.st] : cvbf;
      const bf16_t* mt = msgT + (size_t)((pass == 0) ? e.rm : e.cm) * 8192;
      hgt_edge_msg_wmma<<<groups, 256, 0, stream>>>(vsrc, srcp, dstp, mt, att,
                                                    hacc, e.E);
      hgt_norm_acc<<<(Nd * 256 + 255) / 256, 256, 0, stream>>>(
          hacc, psum, oacc + out_off[e.dt], Nd);
    }
  }

  // Stage 3: mean over etypes -> ReLU -> LayerNorm; outputs word|topic|doc
  const float inv_cnt[3] = {1.f / 3.f, 1.f, 0.5f};
  for (int t = 0; t < 3; ++t)
    hgt_finalize<<<NN_[t], 256, 0, stream>>>(
        oacc + out_off[t], (const float*)d_in[ln_g_idx[t]],
        (const float*)d_in[ln_b_idx[t]], (float*)d_out + out_off[t],
        inv_cnt[t], NN_[t]);
}
